// SentenceLSTM_27358941675745
// MI455X (gfx1250) — compile-verified, matrix operand order checked by reading, not run on hardware
//
#include <hip/hip_runtime.h>
#include <hip/hip_bf16.h>

// ---------------------------------------------------------------------------
// SentenceLSTM forward for MI455X (gfx1250), wave32 + WMMA bf16.
//
// All GEMMs run through v_wmma_f32_16x16x32_bf16 (f32 operands converted to
// bf16 in-register via v_cvt_pk_bf16_f32, f32 accumulation). Each wave owns a
// 32x64 output macro-tile: 2 A-frags x 4 B-frags -> 8 WMMAs per K-step,
// 341 FLOP per loaded byte (vs 128 for a single 16x16 tile).
// Attention phase is HBM-bound (streams v [411MB f32] each of 10 steps) and
// is a coalesced float4 kernel; enc_proj is kept bf16 (51MB, L2-resident on
// the 192MB L2) to halve the tanh-attention read traffic.
// ---------------------------------------------------------------------------

typedef __attribute__((ext_vector_type(16))) __bf16 bf16x16;
typedef __attribute__((ext_vector_type(8)))  float  f32x8;

// ---- problem constants (match reference) ----
#define Bc  256
#define Gc  14
#define Nc  (Gc * Gc)      // 196 spatial positions
#define Dc  2048           // vis embed dim
#define Hc  1024           // lstm hidden
#define Ac  512            // attention dim
#define SIc 1024           // sent input (ctx) dim
#define WIc 1024           // topic dim
#define STc 256            // stop hidden dim
#define Tc  10             // sentences (scan length)

static __device__ __forceinline__ float sigmoidf_(float x) {
  return 1.0f / (1.0f + __expf(-x));
}

// Build a 16x32 bf16 WMMA fragment row (A) or column (B) from an f32 row.
// Per CDNA5 16-bit A/B layout: lanes 0-15 hold K {k0..k0+7, k0+16..k0+23},
// lanes 16-31 hold K {k0+8..k0+15, k0+24..k0+31}. hf = lane>>4.
static __device__ __forceinline__ bf16x16 frag_from_f32(const float* __restrict__ p,
                                                        int k0, int hf) {
  const float* g0 = p + k0 + (hf << 3);
  const float* g1 = g0 + 16;
  bf16x16 r;
#pragma unroll
  for (int i = 0; i < 8; ++i) r[i] = (__bf16)g0[i];
#pragma unroll
  for (int i = 0; i < 8; ++i) r[8 + i] = (__bf16)g1[i];
  return r;
}

static __device__ __forceinline__ f32x8 wmma_bf16(bf16x16 a, bf16x16 b, f32x8 c) {
  return __builtin_amdgcn_wmma_f32_16x16x32_bf16(false, a, false, b, (short)0,
                                                 c, false, false);
}

// ---------------------------------------------------------------------------
// Generic fused dual-GEMM, one wave per 32x64 output macro-tile:
//   out = act( A1[M,K1] @ W1[N,K1]^T  (+ A2[M,K2] @ W2[N,K2]^T)  + b1 (+ b2) )
// act: 0 = none, 1 = tanh.  Stores f32 and/or bf16.
// grid = (N/64, M/32), block = 32 (one wave; EXEC all ones for WMMA).
// ---------------------------------------------------------------------------
__global__ void __launch_bounds__(32) k_gemm_dual_wmma(
    const float* __restrict__ A1, const float* __restrict__ W1,
    const float* __restrict__ B1, int K1,
    const float* __restrict__ A2, const float* __restrict__ W2,
    const float* __restrict__ B2, int K2,
    float* __restrict__ outF, __bf16* __restrict__ outBF, int ldO, int act) {
  const int lane = threadIdx.x;
  const int hf   = lane >> 4;
  const int l16  = lane & 15;
  const long m0  = (long)blockIdx.y * 32;   // row base of macro-tile
  const long n0  = (long)blockIdx.x * 64;   // col base of macro-tile

  f32x8 acc[2][4] = {};

  // ---- pass 1: A1 @ W1^T ----
  {
    const float* ar0 = A1 + (m0 + l16) * (long)K1;
    const float* ar1 = A1 + (m0 + 16 + l16) * (long)K1;
    const float* wr0 = W1 + (n0 + l16) * (long)K1;
    const float* wr1 = wr0 + 16 * (long)K1;
    const float* wr2 = wr0 + 32 * (long)K1;
    const float* wr3 = wr0 + 48 * (long)K1;
    for (int k0 = 0; k0 < K1; k0 += 32) {
      const bf16x16 a0 = frag_from_f32(ar0, k0, hf);
      const bf16x16 a1 = frag_from_f32(ar1, k0, hf);
      {
        const bf16x16 bb = frag_from_f32(wr0, k0, hf);
        acc[0][0] = wmma_bf16(a0, bb, acc[0][0]);
        acc[1][0] = wmma_bf16(a1, bb, acc[1][0]);
      }
      {
        const bf16x16 bb = frag_from_f32(wr1, k0, hf);
        acc[0][1] = wmma_bf16(a0, bb, acc[0][1]);
        acc[1][1] = wmma_bf16(a1, bb, acc[1][1]);
      }
      {
        const bf16x16 bb = frag_from_f32(wr2, k0, hf);
        acc[0][2] = wmma_bf16(a0, bb, acc[0][2]);
        acc[1][2] = wmma_bf16(a1, bb, acc[1][2]);
      }
      {
        const bf16x16 bb = frag_from_f32(wr3, k0, hf);
        acc[0][3] = wmma_bf16(a0, bb, acc[0][3]);
        acc[1][3] = wmma_bf16(a1, bb, acc[1][3]);
      }
    }
  }

  // ---- optional pass 2: + A2 @ W2^T ----
  if (W2) {
    const float* ar0 = A2 + (m0 + l16) * (long)K2;
    const float* ar1 = A2 + (m0 + 16 + l16) * (long)K2;
    const float* wr0 = W2 + (n0 + l16) * (long)K2;
    const float* wr1 = wr0 + 16 * (long)K2;
    const float* wr2 = wr0 + 32 * (long)K2;
    const float* wr3 = wr0 + 48 * (long)K2;
    for (int k0 = 0; k0 < K2; k0 += 32) {
      const bf16x16 a0 = frag_from_f32(ar0, k0, hf);
      const bf16x16 a1 = frag_from_f32(ar1, k0, hf);
      {
        const bf16x16 bb = frag_from_f32(wr0, k0, hf);
        acc[0][0] = wmma_bf16(a0, bb, acc[0][0]);
        acc[1][0] = wmma_bf16(a1, bb, acc[1][0]);
      }
      {
        const bf16x16 bb = frag_from_f32(wr1, k0, hf);
        acc[0][1] = wmma_bf16(a0, bb, acc[0][1]);
        acc[1][1] = wmma_bf16(a1, bb, acc[1][1]);
      }
      {
        const bf16x16 bb = frag_from_f32(wr2, k0, hf);
        acc[0][2] = wmma_bf16(a0, bb, acc[0][2]);
        acc[1][2] = wmma_bf16(a1, bb, acc[1][2]);
      }
      {
        const bf16x16 bb = frag_from_f32(wr3, k0, hf);
        acc[0][3] = wmma_bf16(a0, bb, acc[0][3]);
        acc[1][3] = wmma_bf16(a1, bb, acc[1][3]);
      }
    }
  }

  // ---- epilogue ----
  // C layout per 16x16 tile: VGPR r -> row = base + hf*8 + r, col = base + l16
#pragma unroll
  for (int j = 0; j < 4; ++j) {
    const int col = (int)n0 + j * 16 + l16;
    float bias = 0.0f;
    if (B1) bias += B1[col];
    if (B2) bias += B2[col];
#pragma unroll
    for (int mi = 0; mi < 2; ++mi) {
      const long rbase = m0 + mi * 16 + hf * 8;
#pragma unroll
      for (int r = 0; r < 8; ++r) {
        float v = acc[mi][j][r] + bias;
        if (act == 1) v = tanhf(v);
        const long o = (rbase + r) * (long)ldO + col;
        if (outF)  outF[o]  = v;
        if (outBF) outBF[o] = (__bf16)v;
      }
    }
  }
}

// ---------------------------------------------------------------------------
// Attention step: per batch b,
//   logit[n] = fab + sum_a faw[a] * tanh(encp[b,n,a] + dec[b,a])
//   prob     = softmax(logit) ;  att_out[b,d] = sum_n prob[n] * v[b,n,d]
// One block (256 thr, 8 waves) per batch element. Phase 3 streams v (HBM-bound).
// ---------------------------------------------------------------------------
__global__ void __launch_bounds__(256) k_attention(
    const __bf16* __restrict__ encp,   // [B, N, A] bf16
    const float* __restrict__ dec,     // [B, A]
    const float* __restrict__ vis,     // [B, N, D]
    const float* __restrict__ faw,     // [A]
    const float* __restrict__ fab,     // [1]
    float* __restrict__ attout)        // [B, D]
{
  __shared__ float s_dec[Ac];
  __shared__ float s_faw[Ac];
  __shared__ float s_prob[Nc + 4];

  const int b   = blockIdx.x;
  const int tid = threadIdx.x;
  const int wv  = tid >> 5;
  const int ln  = tid & 31;

  for (int i = tid; i < Ac; i += 256) {
    s_dec[i] = dec[(long)b * Ac + i];
    s_faw[i] = faw[i];
  }
  __syncthreads();

  // Phase 1: logits (one wave per n, round-robin)
  for (int n = wv; n < Nc; n += 8) {
    const __bf16* ep = encp + ((long)b * Nc + n) * Ac;
    float s = 0.0f;
    for (int a = ln; a < Ac; a += 32)
      s += s_faw[a] * tanhf((float)ep[a] + s_dec[a]);
#pragma unroll
    for (int off = 16; off > 0; off >>= 1) s += __shfl_xor(s, off, 32);
    if (ln == 0) s_prob[n] = s + fab[0];
  }
  __syncthreads();

  // Phase 2: softmax over Nc (wave 0)
  if (wv == 0) {
    float m = -3.0e38f;
    for (int n = ln; n < Nc; n += 32) m = fmaxf(m, s_prob[n]);
#pragma unroll
    for (int off = 16; off > 0; off >>= 1) m = fmaxf(m, __shfl_xor(m, off, 32));
    float sum = 0.0f;
    for (int n = ln; n < Nc; n += 32) {
      float e = __expf(s_prob[n] - m);
      s_prob[n] = e;
      sum += e;
    }
#pragma unroll
    for (int off = 16; off > 0; off >>= 1) sum += __shfl_xor(sum, off, 32);
    const float inv = 1.0f / sum;
    for (int n = ln; n < Nc; n += 32) s_prob[n] *= inv;
  }
  __syncthreads();

  // Phase 3: att_out = prob @ v   (coalesced float4 streaming of v)
  const float* vb = vis + (long)b * Nc * Dc;
  for (int d4 = tid * 4; d4 < Dc; d4 += 256 * 4) {
    float4 acc = make_float4(0.f, 0.f, 0.f, 0.f);
    for (int n = 0; n < Nc; ++n) {
      __builtin_prefetch(vb + (long)(n + 2) * Dc + d4, 0, 1);
      const float4 x = *(const float4*)(vb + (long)n * Dc + d4);
      const float p = s_prob[n];
      acc.x += p * x.x; acc.y += p * x.y; acc.z += p * x.z; acc.w += p * x.w;
    }
    *(float4*)(attout + (long)b * Dc + d4) = acc;
  }
}

// ---------------------------------------------------------------------------
// LSTM pointwise: gates [B,4H] (i,f,g,o) -> c,h update.
// ---------------------------------------------------------------------------
__global__ void __launch_bounds__(256) k_lstm(
    const float* __restrict__ gates, const float* __restrict__ c_in,
    float* __restrict__ c_out, float* __restrict__ h_out) {
  const long idx = (long)blockIdx.x * 256 + threadIdx.x;   // B*H total
  const int b = (int)(idx >> 10);
  const int j = (int)(idx & (Hc - 1));
  const float* g = gates + (long)b * 4 * Hc;
  const float gi = sigmoidf_(g[j]);
  const float gf = sigmoidf_(g[Hc + j]);
  const float gg = tanhf(g[2 * Hc + j]);
  const float go = sigmoidf_(g[3 * Hc + j]);
  const float cn = gf * c_in[idx] + gi * gg;
  c_out[idx] = cn;
  h_out[idx] = go * tanhf(cn);
}

// ---------------------------------------------------------------------------
// Tiny final head: p[b, t, j] = final_b[j] + pbuf[b,:] . final_w[j,:]
// ---------------------------------------------------------------------------
__global__ void __launch_bounds__(256) k_final(
    const float* __restrict__ pbuf, const float* __restrict__ fw,
    const float* __restrict__ fb, float* __restrict__ outp, int t) {
  const int idx = blockIdx.x * 256 + threadIdx.x;  // 512 total
  if (idx >= Bc * 2) return;
  const int b = idx >> 1;
  const int j = idx & 1;
  const float* p = pbuf + (long)b * STc;
  const float* w = fw + (long)j * STc;
  float s = fb[j];
  for (int k = 0; k < STc; ++k) s += p[k] * w[k];
  outp[((long)b * Tc + t) * 2 + j] = s;
}

__global__ void __launch_bounds__(256) k_zero(float* __restrict__ p, long n) {
  const long i = (long)blockIdx.x * 256 + threadIdx.x;
  if (i < n) p[i] = 0.0f;
}

// ---------------------------------------------------------------------------
// Host orchestration.
// ---------------------------------------------------------------------------
extern "C" void kernel_launch(void* const* d_in, const int* in_sizes, int n_in,
                              void* d_out, int out_size, void* d_ws, size_t ws_size,
                              hipStream_t stream) {
  (void)in_sizes; (void)n_in; (void)out_size; (void)ws_size;

  const float* vis        = (const float*)d_in[0];   // [B,G,G,D] == [B,N,D]
  // d_in[1] = captions (only its length T matters; constant 10)
  const float* enc_att_w  = (const float*)d_in[2];
  const float* enc_att_b  = (const float*)d_in[3];
  const float* dec_att_w  = (const float*)d_in[4];
  const float* dec_att_b  = (const float*)d_in[5];
  const float* full_att_w = (const float*)d_in[6];
  const float* full_att_b = (const float*)d_in[7];
  const float* ctx_w      = (const float*)d_in[8];
  const float* ctx_b      = (const float*)d_in[9];
  const float* w_ih       = (const float*)d_in[10];
  const float* b_ih       = (const float*)d_in[11];
  const float* w_hh       = (const float*)d_in[12];
  const float* b_hh       = (const float*)d_in[13];
  const float* topic_h_w  = (const float*)d_in[14];
  const float* topic_h_b  = (const float*)d_in[15];
  const float* topic_c_w  = (const float*)d_in[16];
  const float* topic_c_b  = (const float*)d_in[17];
  const float* stop_p_w   = (const float*)d_in[18];
  const float* stop_p_b   = (const float*)d_in[19];
  const float* stop_c_w   = (const float*)d_in[20];
  const float* stop_c_b   = (const float*)d_in[21];
  const float* final_w    = (const float*)d_in[22];
  const float* final_b    = (const float*)d_in[23];

  float* out   = (float*)d_out;
  float* out_t = out;                              // topics  [B, T, WI]
  float* out_p = out + (long)Bc * Tc * WIc;        // ps      [B, T, 2]

  // ---- carve workspace ----
  char*  base = (char*)d_ws;
  size_t off  = 0;
  auto carve = [&](size_t bytes) -> void* {
    off = (off + 255) & ~(size_t)255;
    void* p = base + off;
    off += bytes;
    return p;
  };
  __bf16* encp   = (__bf16*)carve((size_t)Bc * Nc * Ac * sizeof(__bf16)); // 51.4 MB
  float*  dec    = (float*) carve((size_t)Bc * Ac  * sizeof(float));
  float*  attout = (float*) carve((size_t)Bc * Dc  * sizeof(float));
  float*  ctxb   = (float*) carve((size_t)Bc * SIc * sizeof(float));
  float*  gates  = (float*) carve((size_t)Bc * 4 * Hc * sizeof(float));
  float*  hA     = (float*) carve((size_t)Bc * Hc  * sizeof(float));
  float*  hB     = (float*) carve((size_t)Bc * Hc  * sizeof(float));
  float*  cbuf   = (float*) carve((size_t)Bc * Hc  * sizeof(float));
  float*  pbuf   = (float*) carve((size_t)Bc * STc * sizeof(float));

  auto gemm = [&](const float* A1, const float* W1, const float* B1, int K1,
                  const float* A2, const float* W2, const float* B2, int K2,
                  float* oF, __bf16* oBF, int M, int N, int ldO, int act) {
    dim3 grid(N / 64, M / 32);
    k_gemm_dual_wmma<<<grid, 32, 0, stream>>>(A1, W1, B1, K1, A2, W2, B2, K2,
                                              oF, oBF, ldO, act);
  };

  // h0 = c0 = 0
  {
    const long n = (long)Bc * Hc;
    k_zero<<<(int)((n + 255) / 256), 256, 0, stream>>>(hA, n);
    k_zero<<<(int)((n + 255) / 256), 256, 0, stream>>>(cbuf, n);
  }

  // enc_proj = v @ enc_att_w^T + b  -> bf16 [B*N, A]   (hoisted, the 105-GFLOP GEMM)
  gemm(vis, enc_att_w, enc_att_b, Dc, nullptr, nullptr, nullptr, 0,
       nullptr, encp, Bc * Nc, Ac, Ac, 0);

  float* hcur = hA;   // h_{t}
  float* hnew = hB;   // h_{t+1}

  for (int t = 0; t < Tc; ++t) {
    // dec = h @ dec_att_w^T + b          [B, A]
    gemm(hcur, dec_att_w, dec_att_b, Hc, nullptr, nullptr, nullptr, 0,
         dec, nullptr, Bc, Ac, Ac, 0);

    // attention -> att_out [B, D]
    k_attention<<<Bc, 256, 0, stream>>>(encp, dec, vis, full_att_w, full_att_b,
                                        attout);

    // ctx = att_out @ ctx_w^T + b        [B, SI]
    gemm(attout, ctx_w, ctx_b, Dc, nullptr, nullptr, nullptr, 0,
         ctxb, nullptr, Bc, SIc, SIc, 0);

    // gates = ctx @ w_ih^T + b_ih + h @ w_hh^T + b_hh   [B, 4H]
    gemm(ctxb, w_ih, b_ih, SIc, hcur, w_hh, b_hh, Hc,
         gates, nullptr, Bc, 4 * Hc, 4 * Hc, 0);

    // LSTM pointwise -> c (in place), h_new
    k_lstm<<<(Bc * Hc) / 256, 256, 0, stream>>>(gates, cbuf, cbuf, hnew);

    // topic = tanh(h_new @ topic_h^T + b + ctx @ topic_c^T + b) -> out[:, t, :]
    gemm(hnew, topic_h_w, topic_h_b, Hc, ctxb, topic_c_w, topic_c_b, SIc,
         out_t + (long)t * WIc, nullptr, Bc, WIc, Tc * WIc, 1);

    // pbuf = tanh(h_prev @ stop_p^T + b + h_new @ stop_c^T + b)  [B, ST]
    gemm(hcur, stop_p_w, stop_p_b, Hc, hnew, stop_c_w, stop_c_b, Hc,
         pbuf, nullptr, Bc, STc, STc, 1);

    // p = pbuf @ final^T + b -> out_p[:, t, :]
    k_final<<<2, 256, 0, stream>>>(pbuf, final_w, final_b, out_p, t);

    // ping-pong h
    float* tmp = hcur; hcur = hnew; hnew = tmp;
  }
}
